// LoopModel_58815282152080
// MI455X (gfx1250) — matrix operand order repaired and checked by reading, not compiled, not run on hardware
//
#include <hip/hip_runtime.h>
#include <math.h>

typedef __bf16 bf16;
typedef __attribute__((ext_vector_type(16))) __bf16 v16bf;
typedef __attribute__((ext_vector_type(8)))  __bf16 v8bf;
typedef __attribute__((ext_vector_type(8)))  float  v8f;

#define B_     1024
#define L_     128
#define W_     512
#define S_     8
#define STEPS_ 12
#define NLAB_  10

// ---------- small helpers ----------
__device__ __forceinline__ bf16 f2bf(float f) {
  unsigned u; __builtin_memcpy(&u, &f, 4);
  unsigned r = u + 0x7FFFu + ((u >> 16) & 1u);     // round-to-nearest-even
  unsigned short h = (unsigned short)(r >> 16);
  bf16 o; __builtin_memcpy(&o, &h, 2); return o;
}
__device__ __forceinline__ float bf2f(bf16 b) {
  unsigned short h; __builtin_memcpy(&h, &b, 2);
  unsigned u = ((unsigned)h) << 16;
  float f; __builtin_memcpy(&f, &u, 4); return f;
}
__device__ __forceinline__ float sigmoidf_(float x) { return 1.0f / (1.0f + expf(-x)); }

// ---------- fp32 -> bf16 weight conversion ----------
__global__ void f2bf_kernel(const float* __restrict__ src, bf16* __restrict__ dst, long n) {
  long i = (long)blockIdx.x * blockDim.x + threadIdx.x;
  long stride = (long)gridDim.x * blockDim.x;
  for (; i < n; i += stride) dst[i] = f2bf(src[i]);
}

// ---------- encoder: masked mean / max / last -> cat_bf [B, 3W] (bf16) ----------
__global__ __launch_bounds__(256) void encode_kernel(
    const int* __restrict__ x, const float* __restrict__ embed, bf16* __restrict__ cat_bf) {
  int b = blockIdx.x, tid = threadIdx.x;
  int count = 0;
  for (int l = 0; l < L_; ++l) count += (x[b * L_ + l] != 0);
  int last_idx = count > 0 ? count - 1 : 0;
  int last_tok = x[b * L_ + last_idx];
  float denom = count > 0 ? (float)count : 1.0f;
  for (int w = tid; w < W_; w += 256) {
    float sum = 0.0f, mx = -10000.0f;
    for (int l = 0; l < L_; ++l) {
      int tok = x[b * L_ + l];
      if (tok != 0) {
        float e = embed[tok * W_ + w];
        sum += e;
        mx = fmaxf(mx, e);
      }
    }
    long cb = (long)b * (3 * W_);
    cat_bf[cb + w]            = f2bf(sum / denom);
    cat_bf[cb + W_ + w]       = f2bf(mx);
    cat_bf[cb + 2 * W_ + w]   = f2bf(embed[last_tok * W_ + w]);
  }
}

// ---------- batched bf16 WMMA GEMM with async-LDS double buffering ----------
// C[m,n] = act( sum_k A[m,k]*Wt[n,k] + bias[n] ...)
// Block: 256 threads = 8 wave32.  Block tile 128x128, K-step 32.
// K-tiles staged with GLOBAL_LOAD_ASYNC_TO_LDS_B128 (ASYNCcnt) into 2 ping-pong
// LDS buffers; DMA of tile it+1 overlaps WMMA compute of tile it.
// Wave grid 4(M) x 2(N); wave tile 32x64 -> 2x4 fragments, 8 WMMA per K-step.
#define TM 128
#define TN 128
#define TK 32
#define LSTR 40   // LDS row stride in halves (32 + 8 pad), keeps 16B alignment

__global__ __launch_bounds__(256) void wmma_gemm_bf16(
    const bf16* __restrict__ A, long strideAb, int ldA, int aRot,
    const bf16* __restrict__ Wt, long strideWb,
    const float* __restrict__ bias, long strideBiasb,
    float* __restrict__ Cf, bf16* __restrict__ Cb, long strideCb, int ldC,
    const bf16* __restrict__ Add, long strideAddb, int ldAdd,
    const float* __restrict__ gate,
    float* __restrict__ h_init, bf16* __restrict__ shared_out,
    int K, int Sn, int mode) {
  __shared__ __align__(16) bf16 As[2 * TM * LSTR];
  __shared__ __align__(16) bf16 Bs[2 * TN * LSTR];

  const int tid  = threadIdx.x;
  const int s    = blockIdx.z;
  const int aBatch = (s + aRot) % Sn;                 // ring-roll of A over batches
  const bf16*  Ab    = A    + (long)aBatch * strideAb;
  const bf16*  Wb    = Wt   + (long)s * strideWb;
  const float* biasb = bias + (long)s * strideBiasb;

  const int m0 = blockIdx.y * TM;
  const int n0 = blockIdx.x * TN;

  const int wave = tid >> 5;
  const int lane = tid & 31;
  const int wm = wave & 3;     // 4 waves along M
  const int wn = wave >> 2;    // 2 waves along N

  v8f acc[2][4];
  #pragma unroll
  for (int i = 0; i < 2; ++i)
    #pragma unroll
    for (int j = 0; j < 4; ++j)
      #pragma unroll
      for (int r = 0; r < 8; ++r) acc[i][j][r] = 0.0f;

  // per-thread slice of the K-tile: row lrow, halves [lkh, lkh+16)
  const int lrow = tid >> 1;           // 0..127
  const int lkh  = (tid & 1) * 16;     // 0 or 16 halves
  const bf16* gArow = Ab + (long)(m0 + lrow) * ldA + lkh;
  const bf16* gBrow = Wb + (long)(n0 + lrow) * K + lkh;
  // LDS byte address = low 32 bits of the generic pointer (local offset)
  const unsigned ldsA0 = (unsigned)(unsigned long long)&As[lrow * LSTR + lkh];
  const unsigned ldsB0 = (unsigned)(unsigned long long)&Bs[lrow * LSTR + lkh];
  const unsigned bufAstride = (unsigned)(TM * LSTR * sizeof(bf16));
  const unsigned bufBstride = (unsigned)(TN * LSTR * sizeof(bf16));

  auto issue_tile = [&](int k0, int buf) {
    unsigned la = ldsA0 + (unsigned)buf * bufAstride;
    unsigned lb = ldsB0 + (unsigned)buf * bufBstride;
    unsigned long long ga = (unsigned long long)(const void*)(gArow + k0);
    unsigned long long gb = (unsigned long long)(const void*)(gBrow + k0);
    // 16B per op; IOFFSET applies to both the global and the LDS address.
    asm volatile("global_load_async_to_lds_b128 %0, %1, off"
                 :: "v"(la), "v"(ga) : "memory");
    asm volatile("global_load_async_to_lds_b128 %0, %1, off offset:16"
                 :: "v"(la), "v"(ga) : "memory");
    asm volatile("global_load_async_to_lds_b128 %0, %1, off"
                 :: "v"(lb), "v"(gb) : "memory");
    asm volatile("global_load_async_to_lds_b128 %0, %1, off offset:16"
                 :: "v"(lb), "v"(gb) : "memory");
  };

  // ISA lane layouts (CDNA5 ISA):
  //  A 16x32 bf16: lane<16 -> M=lane, K in {kb..kb+7, kb+16..kb+23}, kb=(lane>>4)*8
  //  B 32x16 bf16: lane<16 -> N=lane, K in {kb..kb+7, kb+16..kb+23}, kb=(lane>>4)*8
  const int kb = (lane >> 4) * 8;
  const int ko = (lane >> 4) * 16;
  const int ml = lane & 15;

  const int nk = K / TK;
  issue_tile(0, 0);
  for (int it = 0; it < nk; ++it) {
    const int cur = it & 1;
    // tile 'it' is the only set of asyncs this wave has in flight here
    asm volatile("s_wait_asynccnt 0x0" ::: "memory");
    __syncthreads();   // everyone's DMA for tile 'it' landed; prev readers of cur^1 done
    if (it + 1 < nk) issue_tile((it + 1) * TK, cur ^ 1);  // overlaps compute below

    const bf16* Asb = &As[cur * TM * LSTR];
    const bf16* Bsb = &Bs[cur * TN * LSTR];

    v16bf af[2];
    #pragma unroll
    for (int i = 0; i < 2; ++i) {
      const bf16* p = &Asb[(wm * 32 + i * 16 + ml) * LSTR + kb];
      v8bf lo = *(const v8bf*)p;
      v8bf hi = *(const v8bf*)(p + 16);
      #pragma unroll
      for (int e = 0; e < 8; ++e) { af[i][e] = lo[e]; af[i][e + 8] = hi[e]; }
    }
    v16bf bv[4];
    #pragma unroll
    for (int j = 0; j < 4; ++j) {
      const bf16* p = &Bsb[(wn * 64 + j * 16 + ml) * LSTR + kb];
      v8bf lo = *(const v8bf*)p;
      v8bf hi = *(const v8bf*)(p + 16);
      #pragma unroll
      for (int e = 0; e < 8; ++e) { bv[j][e] = lo[e]; bv[j][e + 8] = hi[e]; }
    }
    #pragma unroll
    for (int i = 0; i < 2; ++i)
      #pragma unroll
      for (int j = 0; j < 4; ++j)
        acc[i][j] = __builtin_amdgcn_wmma_f32_16x16x32_bf16(
            false, af[i], false, bv[j], (short)0, acc[i][j], false, false);
  }

  // C/D layout: VGPR r -> M = r (lanes 0-15) / r+8 (lanes 16-31), N = lane&15
  const int lm = (lane < 16) ? 0 : 8;
  #pragma unroll
  for (int i = 0; i < 2; ++i) {
    #pragma unroll
    for (int j = 0; j < 4; ++j) {
      int col = n0 + wn * 64 + j * 16 + ml;
      float bvv = biasb[col];
      #pragma unroll
      for (int r = 0; r < 8; ++r) {
        int row = m0 + wm * 32 + i * 16 + lm + r;
        float v = acc[i][j][r] + bvv;
        long cidx = (long)row * ldC + (long)s * strideCb + col;
        if (mode == 0) {                 // encode: summary -> h init + shared=[h|summary]
          float t = tanhf(v);
          bf16 tb = f2bf(t);
          h_init[(long)row * W_ + col] = t;
          shared_out[(long)row * (2 * W_) + col]       = tb;
          shared_out[(long)row * (2 * W_) + W_ + col]  = tb;
        } else if (mode == 1) {          // tanh -> bf16  (f0)
          Cb[cidx] = f2bf(tanhf(v));
        } else if (mode == 2) {          // tanh(v + f0) -> bf16  (feats)
          float a = bf2f(Add[(long)row * ldAdd + (long)s * strideAddb + col]);
          Cb[cidx] = f2bf(tanhf(v + a));
        } else if (mode == 3) {          // bias only -> f32  (gi / gh)
          Cf[cidx] = v;
        } else {                         // gate * tanh -> f32  (ups)
          float g = gate[(long)row * Sn + s];
          Cf[cidx] = g * tanhf(v);
        }
      }
    }
  }
}

// ---------- fused GRU cell + gate reduction + prismion ----------
__global__ __launch_bounds__(256) void gru_kernel(
    const float* __restrict__ gi, const float* __restrict__ gh,
    float* __restrict__ hs, bf16* __restrict__ hs_bf,
    const float* __restrict__ Wgate, const float* __restrict__ bgate,
    const float* __restrict__ phase, const float* __restrict__ pgain,
    bf16* __restrict__ feat2_bf, float* __restrict__ gate) {
  int b = blockIdx.x, s = blockIdx.y, tid = threadIdx.x;
  long base3 = ((long)b * S_ + s) * (3 * W_);
  long baseW = ((long)b * S_ + s) * W_;
  float gpart = 0.0f;
  for (int w = tid; w < W_; w += 256) {
    float ir  = gi[base3 + w], iz = gi[base3 + W_ + w], inn = gi[base3 + 2 * W_ + w];
    float hr  = gh[base3 + w], hz = gh[base3 + W_ + w], hn  = gh[base3 + 2 * W_ + w];
    float hp  = hs[baseW + w];
    float r = sigmoidf_(ir + hr);
    float z = sigmoidf_(iz + hz);
    float n = tanhf(inn + r * hn);
    float hnew = (1.0f - z) * n + z * hp;
    hs[baseW + w] = hnew;
    hs_bf[baseW + w] = f2bf(hnew);
    gpart += Wgate[s * W_ + w] * hnew;
    float c = cosf(hnew + phase[s * W_ + w]);
    float gx = pgain[s * W_ + w];
    float gn = (gx > 20.0f) ? gx : log1pf(expf(gx));  // softplus
    feat2_bf[baseW + w] = f2bf(c * c * gn);
  }
  __shared__ float red[256];
  red[tid] = gpart;
  __syncthreads();
  for (int off = 128; off > 0; off >>= 1) {
    if (tid < off) red[tid] += red[tid + off];
    __syncthreads();
  }
  if (tid == 0) gate[(long)b * S_ + s] = sigmoidf_(red[0] + bgate[s]);
}

// ---------- h update: delta sum over S, decay, tanh; refresh bf16 shared[:,0:W] ----------
__global__ __launch_bounds__(256) void hupdate_kernel(
    const float* __restrict__ ups, float* __restrict__ h,
    bf16* __restrict__ shared_bf, const float* __restrict__ decay_param) {
  int idx = blockIdx.x * 256 + threadIdx.x;
  int b = idx >> 9, w = idx & 511;
  float d = 0.0f;
  #pragma unroll
  for (int s = 0; s < S_; ++s) d += ups[((long)b * S_ + s) * W_ + w];
  d *= 0.35355339059327373f;   // 1/sqrt(8)
  float decay = sigmoidf_(decay_param[0]);
  float hn = tanhf(h[idx] * decay + d);
  h[idx] = hn;
  shared_bf[(long)b * (2 * W_) + w] = f2bf(hn);
}

// ---------- final projection [B,512] x [10,512]^T ----------
__global__ __launch_bounds__(256) void out_kernel(
    const float* __restrict__ h, const float* __restrict__ Wout,
    const float* __restrict__ bout, float* __restrict__ out) {
  int b = blockIdx.x, tid = threadIdx.x;
  __shared__ float red[256];
  float p[NLAB_];
  #pragma unroll
  for (int n = 0; n < NLAB_; ++n) p[n] = 0.0f;
  for (int w = tid; w < W_; w += 256) {
    float hv = h[(long)b * W_ + w];
    #pragma unroll
    for (int n = 0; n < NLAB_; ++n) p[n] += hv * Wout[n * W_ + w];
  }
  for (int n = 0; n < NLAB_; ++n) {
    red[tid] = p[n];
    __syncthreads();
    for (int off = 128; off > 0; off >>= 1) {
      if (tid < off) red[tid] += red[tid + off];
      __syncthreads();
    }
    if (tid == 0) out[(long)b * NLAB_ + n] = red[0] + bout[n];
    __syncthreads();
  }
}

// ---------- host side ----------
extern "C" void kernel_launch(void* const* d_in, const int* in_sizes, int n_in,
                              void* d_out, int out_size, void* d_ws, size_t ws_size,
                              hipStream_t stream) {
  (void)in_sizes; (void)n_in; (void)out_size; (void)ws_size;
  const int*   x       = (const int*)  d_in[0];
  const float* embed   = (const float*)d_in[1];
  const float* W_mix   = (const float*)d_in[2];
  const float* b_mix   = (const float*)d_in[3];
  const float* W_in    = (const float*)d_in[4];
  const float* b_in    = (const float*)d_in[5];
  const float* W_link  = (const float*)d_in[6];
  const float* b_link  = (const float*)d_in[7];
  const float* W_ih    = (const float*)d_in[8];
  const float* b_ih    = (const float*)d_in[9];
  const float* W_hh    = (const float*)d_in[10];
  const float* b_hh    = (const float*)d_in[11];
  const float* W_gate  = (const float*)d_in[12];
  const float* b_gate  = (const float*)d_in[13];
  const float* phase   = (const float*)d_in[14];
  const float* pgain   = (const float*)d_in[15];
  const float* W_delta = (const float*)d_in[16];
  const float* b_delta = (const float*)d_in[17];
  const float* decay   = (const float*)d_in[18];
  const float* W_out   = (const float*)d_in[19];
  const float* b_out   = (const float*)d_in[20];
  float* out = (float*)d_out;

  char* base = (char*)d_ws;
  size_t off = 0;
  auto alloc = [&](size_t bytes) -> void* {
    void* p = base + off;
    off = (off + bytes + 255) & ~(size_t)255;
    return p;
  };

  const long nWmix  = (long)W_ * 3 * W_;          // 786432
  const long nWin   = (long)S_ * W_ * 2 * W_;     // 4194304
  const long nWlink = (long)S_ * W_ * W_;         // 2097152
  const long nWih   = (long)S_ * 3 * W_ * W_;     // 6291456
  const long nWdel  = (long)S_ * W_ * W_;         // 2097152

  bf16* Wmix_bf   = (bf16*)alloc(nWmix  * 2);
  bf16* Win_bf    = (bf16*)alloc(nWin   * 2);
  bf16* Wlink_bf  = (bf16*)alloc(nWlink * 2);
  bf16* Wih_bf    = (bf16*)alloc(nWih   * 2);
  bf16* Whh_bf    = (bf16*)alloc(nWih   * 2);
  bf16* Wdel_bf   = (bf16*)alloc(nWdel  * 2);
  bf16* cat_bf    = (bf16*)alloc((size_t)B_ * 3 * W_ * 2);
  bf16* shared_bf = (bf16*)alloc((size_t)B_ * 2 * W_ * 2);
  float* h        = (float*)alloc((size_t)B_ * W_ * 4);
  bf16* f0_bf     = (bf16*)alloc((size_t)B_ * S_ * W_ * 2);
  bf16* feats_bf  = (bf16*)alloc((size_t)B_ * S_ * W_ * 2);
  float* hs       = (float*)alloc((size_t)B_ * S_ * W_ * 4);
  bf16* hs_bf     = (bf16*)alloc((size_t)B_ * S_ * W_ * 2);
  float* gi       = (float*)alloc((size_t)B_ * S_ * 3 * W_ * 4);
  float* gh       = (float*)alloc((size_t)B_ * S_ * 3 * W_ * 4);
  bf16* feat2_bf  = (bf16*)alloc((size_t)B_ * S_ * W_ * 2);
  float* gate     = (float*)alloc((size_t)B_ * S_ * 4);
  float* ups      = (float*)alloc((size_t)B_ * S_ * W_ * 4);

  // zero recurrent state (deterministic each call)
  hipMemsetAsync(hs, 0, (size_t)B_ * S_ * W_ * 4, stream);
  hipMemsetAsync(hs_bf, 0, (size_t)B_ * S_ * W_ * 2, stream);

  // weight conversions to bf16
  auto conv = [&](const float* src, bf16* dst, long n) {
    f2bf_kernel<<<dim3(1024), dim3(256), 0, stream>>>(src, dst, n);
  };
  conv(W_mix, Wmix_bf, nWmix);
  conv(W_in, Win_bf, nWin);
  conv(W_link, Wlink_bf, nWlink);
  conv(W_ih, Wih_bf, nWih);
  conv(W_hh, Whh_bf, nWih);
  conv(W_delta, Wdel_bf, nWdel);

  // encode
  encode_kernel<<<dim3(B_), dim3(256), 0, stream>>>(x, embed, cat_bf);

  // summary = tanh(cat @ W_mix^T + b_mix): seeds h and shared=[h|summary]
  wmma_gemm_bf16<<<dim3(W_ / 128, B_ / 128, 1), dim3(256), 0, stream>>>(
      cat_bf, 0L, 3 * W_, 0, Wmix_bf, 0L, b_mix, 0L,
      nullptr, nullptr, 0L, 0, nullptr, 0L, 0, nullptr,
      h, shared_bf, 3 * W_, 1, 0);

  for (int step = 0; step < STEPS_; ++step) {
    // f0 = tanh(shared @ W_in[s]^T + b_in[s])
    wmma_gemm_bf16<<<dim3(W_ / 128, B_ / 128, S_), dim3(256), 0, stream>>>(
        shared_bf, 0L, 2 * W_, 0, Win_bf, (long)W_ * 2 * W_, b_in, (long)W_,
        nullptr, f0_bf, (long)W_, S_ * W_, nullptr, 0L, 0, nullptr,
        nullptr, nullptr, 2 * W_, S_, 1);
    // feats = tanh(f0[s] + f0[s-1] @ W_link[s]^T + b_link[s])   (ring via aRot=S-1)
    wmma_gemm_bf16<<<dim3(W_ / 128, B_ / 128, S_), dim3(256), 0, stream>>>(
        f0_bf, (long)W_, S_ * W_, S_ - 1, Wlink_bf, (long)W_ * W_, b_link, (long)W_,
        nullptr, feats_bf, (long)W_, S_ * W_, f0_bf, (long)W_, S_ * W_, nullptr,
        nullptr, nullptr, W_, S_, 2);
    // gi = feats @ W_ih[s]^T + b_ih[s]
    wmma_gemm_bf16<<<dim3(3 * W_ / 128, B_ / 128, S_), dim3(256), 0, stream>>>(
        feats_bf, (long)W_, S_ * W_, 0, Wih_bf, (long)3 * W_ * W_, b_ih, (long)3 * W_,
        gi, nullptr, (long)3 * W_, S_ * 3 * W_, nullptr, 0L, 0, nullptr,
        nullptr, nullptr, W_, S_, 3);
    // gh = hs @ W_hh[s]^T + b_hh[s]
    wmma_gemm_bf16<<<dim3(3 * W_ / 128, B_ / 128, S_), dim3(256), 0, stream>>>(
        hs_bf, (long)W_, S_ * W_, 0, Whh_bf, (long)3 * W_ * W_, b_hh, (long)3 * W_,
        gh, nullptr, (long)3 * W_, S_ * 3 * W_, nullptr, 0L, 0, nullptr,
        nullptr, nullptr, W_, S_, 3);
    // GRU + gate scalar + prismion feature
    gru_kernel<<<dim3(B_, S_), dim3(256), 0, stream>>>(
        gi, gh, hs, hs_bf, W_gate, b_gate, phase, pgain, feat2_bf, gate);
    // ups = gate * tanh(feat2 @ W_delta[s]^T + b_delta[s])
    wmma_gemm_bf16<<<dim3(W_ / 128, B_ / 128, S_), dim3(256), 0, stream>>>(
        feat2_bf, (long)W_, S_ * W_, 0, Wdel_bf, (long)W_ * W_, b_delta, (long)W_,
        ups, nullptr, (long)W_, S_ * W_, nullptr, 0L, 0, gate,
        nullptr, nullptr, W_, S_, 4);
    // h = tanh(h*decay + sum_s ups / sqrt(S));  refresh shared[:, :W]
    hupdate_kernel<<<dim3((B_ * W_) / 256), dim3(256), 0, stream>>>(
        ups, h, shared_bf, decay);
  }

  out_kernel<<<dim3(B_), dim3(256), 0, stream>>>(h, W_out, b_out, out);
}
